// GNN_61263413510625
// MI455X (gfx1250) — compile-verified, matrix operand order checked by reading, not run on hardware
//
#include <hip/hip_runtime.h>
#include <cstdint>

typedef __attribute__((ext_vector_type(16))) __bf16 v16bf;
typedef __attribute__((ext_vector_type(8)))  float  v8f;

#define EPS_BN 1e-5f

// CDNA5 async memory->LDS path (guarded: falls back to load+ds_store if the
// toolchain doesn't expose the gfx1250 async builtins).
#if defined(__gfx1250__) && __has_builtin(__builtin_amdgcn_global_load_async_to_lds_b128) && \
    __has_builtin(__builtin_amdgcn_s_wait_asynccnt)
#define USE_ASYNC_LDS 1
#else
#define USE_ASYNC_LDS 0
#endif

// exact pointee type the builtin expects: int __attribute__((vector_size(16)))
typedef int async_i4 __attribute__((vector_size(16)));

__device__ __forceinline__ void stage_b128(const __bf16* g, __bf16* l) {
#if USE_ASYNC_LDS
  __builtin_amdgcn_global_load_async_to_lds_b128(
      (__attribute__((address_space(1))) async_i4*)(uintptr_t)g,
      (__attribute__((address_space(3))) async_i4*)(uintptr_t)l, 0, 0);
#else
  *(uint4*)l = *(const uint4*)g;
#endif
}
__device__ __forceinline__ void stage_wait_some() {   // allow 1 in flight
#if USE_ASYNC_LDS
  __builtin_amdgcn_s_wait_asynccnt(1);
#endif
}
__device__ __forceinline__ void stage_wait_all() {
#if USE_ASYNC_LDS
  __builtin_amdgcn_s_wait_asynccnt(0);
#endif
}

// ---------------- utility kernels ----------------

__global__ void k_cvt_bf16(const float* __restrict__ in, __bf16* __restrict__ out, long long n) {
  long long i = (long long)blockIdx.x * blockDim.x + threadIdx.x;
  if (i < n) out[i] = (__bf16)in[i];
}

// W[K][Fo] f32  ->  Wt[Fo][K] bf16 (transpose + convert, weights are L2-resident)
__global__ void k_cvt_transpose_bf16(const float* __restrict__ W, __bf16* __restrict__ Wt,
                                     int K, int Fo) {
  long long i = (long long)blockIdx.x * blockDim.x + threadIdx.x;
  const long long n = (long long)K * Fo;
  if (i >= n) return;
  const int k = (int)(i / Fo);
  const int c = (int)(i % Fo);
  Wt[(size_t)c * K + k] = (__bf16)W[i];
}

__global__ void k_deg(const long long* __restrict__ dst, float* __restrict__ deg, long long E) {
  long long e = (long long)blockIdx.x * blockDim.x + threadIdx.x;
  if (e < E) atomicAdd(&deg[dst[e]], 1.0f);
}

// scatter-add aggregation: thread = (edge, 4-feature group)
__global__ void k_aggregate(const float* __restrict__ x,
                            const long long* __restrict__ src,
                            const long long* __restrict__ dst,
                            float* __restrict__ agg, long long E, int F) {
  long long e = (long long)blockIdx.x * blockDim.x + threadIdx.x;
  if (e >= E) return;
  const int f = blockIdx.y * 4;
  const long long s = src[e], d = dst[e];
  const float4 v = *(const float4*)(x + (size_t)s * F + f);
  float* o = agg + (size_t)d * F + f;
  atomicAdd(o + 0, v.x);
  atomicAdd(o + 1, v.y);
  atomicAdd(o + 2, v.z);
  atomicAdd(o + 3, v.w);
}

__global__ void k_mean_bf16(const float* __restrict__ agg, const float* __restrict__ deg,
                            __bf16* __restrict__ mean, int N, int F) {
  long long i = (long long)blockIdx.x * blockDim.x + threadIdx.x;
  const long long n = (long long)N * F;
  if (i >= n) return;
  const int row = (int)(i / F);
  float d = deg[row];
  d = d > 1.0f ? d : 1.0f;
  mean[i] = (__bf16)(agg[i] / d);
}

// ---------------- WMMA GEMM ----------------
// C[N,Fo] = A1[N,K]@W1 + A2[N,K]@W2 + bias (W given pre-transposed: Wt[Fo][K]).
// Block: 256 thr = 8 waves; wave w owns rows [blkX*128 + w*16, +16), cols [blkY*64, +64).
// Double-buffered LDS B tiles staged via async global->LDS copies.
// Requires K % 32 == 0, Fo % 64 == 0.

union Frag16 { v16bf v; uint4 u[2]; };

#define LDB 40  // padded row stride (elements): 80B rows -> 16B aligned b128, conflict-free

__global__ __launch_bounds__(256) void k_gemm_wmma(
    const __bf16* __restrict__ A1, const __bf16* __restrict__ Wt1,
    const __bf16* __restrict__ A2, const __bf16* __restrict__ Wt2,
    const float* __restrict__ bias, float* __restrict__ C,
    int N, int K, int Fo, int relu) {
  __shared__ __bf16 Bs[2][64][LDB];

  const int tid  = threadIdx.x;
  const int wave = tid >> 5;
  const int lane = tid & 31;
  const int half = lane >> 4;     // 0: lanes 0-15, 1: lanes 16-31
  const int lm   = lane & 15;
  const int row0 = blockIdx.x * 128 + wave * 16;
  const int col0 = blockIdx.y * 64;

  // staging: each thread copies 16B: row n = tid/4 (0..63), k-group = (tid%4)*8
  const int sn = tid >> 2;
  const int sk = (tid & 3) * 8;

  v8f acc[4] = {};                // 4 x (16x16) tiles along N

  int arow = row0 + lm;
  if (arow >= N) arow = N - 1;    // clamp: keep EXEC all-ones for WMMA; OOB rows not stored

  for (int pass = 0; pass < 2; ++pass) {
    const __bf16* A  = pass ? A2 : A1;
    const __bf16* Wt = pass ? Wt2 : Wt1;
    if (A == nullptr) continue;
    const __bf16* arowp = A + (size_t)arow * K;
    const __bf16* wrowp = Wt + (size_t)(col0 + sn) * K + sk;

    stage_b128(wrowp, &Bs[0][sn][sk]);   // prologue: stage k0=0 into buffer 0

    int buf = 0;
    for (int k0 = 0; k0 < K; k0 += 32) {
      if (k0 + 32 < K) {
        stage_b128(wrowp + k0 + 32, &Bs[buf ^ 1][sn][sk]);  // prefetch next tile
        stage_wait_some();                                  // current tile done, next in flight
      } else {
        stage_wait_all();
      }
      __syncthreads();

      // A fragment: 16x32 bf16. Lanes 0-15 hold K=0..7 / 16..23, lanes 16-31 K=8..15 / 24..31
      Frag16 a;
      a.u[0] = *(const uint4*)(arowp + k0 + half * 8);
      a.u[1] = *(const uint4*)(arowp + k0 + 16 + half * 8);

      #pragma unroll
      for (int j = 0; j < 4; ++j) {
        // B fragment: 32x16 bf16; lane holds column n = j*16+lm, same K pattern as A
        Frag16 b;
        const __bf16* bp = &Bs[buf][j * 16 + lm][half * 8];
        b.u[0] = *(const uint4*)(bp);
        b.u[1] = *(const uint4*)(bp + 16);
        acc[j] = __builtin_amdgcn_wmma_f32_16x16x32_bf16(
            false, a.v, false, b.v, (short)0, acc[j], false, false);
      }
      __syncthreads();   // everyone done reading Bs[buf] before it is restaged
      buf ^= 1;
    }
  }

  // epilogue: C/D layout — VGPR v: lanes 0-15 -> M=v, lanes 16-31 -> M=v+8; N = lane%16
  #pragma unroll
  for (int j = 0; j < 4; ++j) {
    const int col = col0 + j * 16 + lm;
    const float bv = bias ? bias[col] : 0.0f;
    #pragma unroll
    for (int v = 0; v < 8; ++v) {
      const int row = row0 + v + half * 8;
      if (row < N) {
        float r = acc[j][v] + bv;
        if (relu) r = fmaxf(r, 0.0f);
        C[(size_t)row * Fo + col] = r;
      }
    }
  }
}

// ---------------- BatchNorm ----------------

__global__ void k_bn_stats(const float* __restrict__ pre, float* __restrict__ sum,
                           float* __restrict__ sumsq, int N, int Fo) {
  const int col = blockIdx.x * blockDim.x + threadIdx.x;
  if (col >= Fo) return;
  const int r0 = blockIdx.y * 4096;
  int r1 = r0 + 4096; if (r1 > N) r1 = N;
  float s = 0.f, s2 = 0.f;
  for (int r = r0; r < r1; ++r) {
    const float v = pre[(size_t)r * Fo + col];
    s += v; s2 += v * v;
  }
  atomicAdd(&sum[col], s);
  atomicAdd(&sumsq[col], s2);
}

__global__ void k_bn_apply(const float* __restrict__ pre, const float* __restrict__ sum,
                           const float* __restrict__ sumsq, const float* __restrict__ gamma,
                           const float* __restrict__ beta, float* __restrict__ hf,
                           __bf16* __restrict__ hb, int N, int Fo) {
  long long i = (long long)blockIdx.x * blockDim.x + threadIdx.x;
  const long long n = (long long)N * Fo;
  if (i >= n) return;
  const int col = (int)(i % Fo);
  const float invN = 1.0f / (float)N;
  const float mu = sum[col] * invN;
  const float var = sumsq[col] * invN - mu * mu;
  float y = (pre[i] - mu) * rsqrtf(var + EPS_BN) * gamma[col] + beta[col];
  y = fmaxf(y, 0.0f);
  hf[i] = y;
  hb[i] = (__bf16)y;
}

// ---------------- host orchestration ----------------

extern "C" void kernel_launch(void* const* d_in, const int* in_sizes, int n_in,
                              void* d_out, int out_size, void* d_ws, size_t ws_size,
                              hipStream_t stream) {
  const int F_IN = 128, H = 1024;
  const int N = in_sizes[0] / F_IN;
  const long long E = (long long)in_sizes[1] / 2;

  const float* x = (const float*)d_in[0];
  const long long* ei = (const long long*)d_in[1];
  const long long* esrc = ei;
  const long long* edst = ei + E;

  const float *Wl[4], *bl[4], *Wr[4], *ga[4], *be[4];
  for (int l = 0; l < 4; ++l) {
    Wl[l] = (const float*)d_in[2 + 5 * l];
    bl[l] = (const float*)d_in[3 + 5 * l];
    Wr[l] = (const float*)d_in[4 + 5 * l];
    ga[l] = (const float*)d_in[5 + 5 * l];
    be[l] = (const float*)d_in[6 + 5 * l];
  }
  const float* Wfc = (const float*)d_in[22];
  const float* bfc = (const float*)d_in[23];

  const int Kd[4] = {128, 1024, 512, 256};
  const int Fd[4] = {1024, 512, 256, 128};

  // ---- workspace carve ----
  char* p = (char*)d_ws;
  auto carve = [&](size_t bytes) {
    char* r = p;
    p += (bytes + 255) & ~(size_t)255;
    return (void*)r;
  };
  float*  deg = (float*)carve((size_t)N * 4);
  float*  agg = (float*)carve((size_t)N * H * 4);  // reused as GEMM output "pre"
  float*  hfA = (float*)carve((size_t)N * H * 4);
  float*  hfB = (float*)carve((size_t)N * H * 4);
  __bf16* hbA = (__bf16*)carve((size_t)N * H * 2);
  __bf16* hbB = (__bf16*)carve((size_t)N * H * 2);
  __bf16* mbf = (__bf16*)carve((size_t)N * H * 2);
  float*  stats = (float*)carve((size_t)2 * H * 4);
  float*  sum = stats;
  float*  sumsq = stats + H;

  // transposed bf16 weights: Wt[Fo][K]
  __bf16* wbf[9];
  const float* wsrc[9] = {Wl[0], Wr[0], Wl[1], Wr[1], Wl[2], Wr[2], Wl[3], Wr[3], Wfc};
  const int wK[9]  = {128, 128, 1024, 1024, 512, 512, 256, 256, 128};
  const int wFo[9] = {1024, 1024, 512, 512, 256, 256, 128, 128, 1024};
  for (int i = 0; i < 9; ++i) wbf[i] = (__bf16*)carve((size_t)wK[i] * wFo[i] * 2);

  const int T = 256;

  // degrees (recomputed each call: deterministic, stateless)
  (void)hipMemsetAsync(deg, 0, (size_t)N * 4, stream);
  k_deg<<<dim3((unsigned)((E + T - 1) / T)), T, 0, stream>>>(edst, deg, E);

  // weights -> bf16, transposed
  for (int i = 0; i < 9; ++i) {
    const long long n = (long long)wK[i] * wFo[i];
    k_cvt_transpose_bf16<<<dim3((unsigned)((n + T - 1) / T)), T, 0, stream>>>(
        wsrc[i], wbf[i], wK[i], wFo[i]);
  }
  // x -> bf16
  {
    const long long n = (long long)N * F_IN;
    k_cvt_bf16<<<dim3((unsigned)((n + T - 1) / T)), T, 0, stream>>>(x, hbA, n);
  }

  const float*  hin_f = x;
  const __bf16* hin_b = hbA;
  float*  houtf[2] = {hfA, hfB};
  __bf16* houtb[2] = {hbB, hbA};  // layer1 must not clobber x_bf16 in hbA

  for (int l = 0; l < 4; ++l) {
    const int K = Kd[l], Fo = Fd[l];

    (void)hipMemsetAsync(agg, 0, (size_t)N * K * 4, stream);
    k_aggregate<<<dim3((unsigned)((E + T - 1) / T), (unsigned)(K / 4)), T, 0, stream>>>(
        hin_f, esrc, edst, agg, E, K);

    {
      const long long n = (long long)N * K;
      k_mean_bf16<<<dim3((unsigned)((n + T - 1) / T)), T, 0, stream>>>(agg, deg, mbf, N, K);
    }

    float* pre = agg;  // GEMM output overwrites agg (mean already extracted)
    k_gemm_wmma<<<dim3((unsigned)((N + 127) / 128), (unsigned)(Fo / 64)), T, 0, stream>>>(
        mbf, wbf[2 * l], hin_b, wbf[2 * l + 1], bl[l], pre, N, K, Fo, 0);

    (void)hipMemsetAsync(stats, 0, (size_t)2 * H * 4, stream);
    k_bn_stats<<<dim3((unsigned)((Fo + T - 1) / T), (unsigned)((N + 4095) / 4096)), T, 0,
                stream>>>(pre, sum, sumsq, N, Fo);
    {
      const long long n = (long long)N * Fo;
      k_bn_apply<<<dim3((unsigned)((n + T - 1) / T)), T, 0, stream>>>(
          pre, sum, sumsq, ga[l], be[l], houtf[l & 1], houtb[l & 1], N, Fo);
    }
    hin_f = houtf[l & 1];
    hin_b = houtb[l & 1];
  }

  // FC: h4_bf16 @ Wfc + bfc, ReLU, straight to d_out (f32)
  k_gemm_wmma<<<dim3((unsigned)((N + 127) / 128), (unsigned)(H / 64)), T, 0, stream>>>(
      hin_b, wbf[8], nullptr, nullptr, bfc, (float*)d_out, N, 128, H, 1);
}
